// MambaUNet_20822001451662
// MI455X (gfx1250) — matrix operand order repaired and checked by reading, not compiled, not run on hardware
//
#include <hip/hip_runtime.h>
#include <hip/hip_bf16.h>
#include <math.h>

typedef __attribute__((ext_vector_type(16))) _Float16 v16h;
typedef __attribute__((ext_vector_type(8)))  float    v8f;
typedef __attribute__((ext_vector_type(4)))  unsigned int u32x4;
typedef __attribute__((ext_vector_type(4)))  int      i32x4;
typedef __attribute__((ext_vector_type(8)))  int      i32x8;
typedef int v4i_ __attribute__((vector_size(16)));   // matches builtin V4i

#define EPSV 1e-5f

// ---- gfx1250 feature probes (compile-safe: absent builtin -> fallback) ----
#if defined(__HIP_DEVICE_COMPILE__) && __has_builtin(__builtin_amdgcn_global_load_async_to_lds_b128) && __has_builtin(__builtin_amdgcn_s_wait_asynccnt)
#define HAVE_ASYNC 1
#define ASYNC_B128(gp, lp)                                                    \
  __builtin_amdgcn_global_load_async_to_lds_b128(                             \
      (__attribute__((address_space(1))) v4i_*)(uintptr_t)(gp),               \
      (__attribute__((address_space(3))) v4i_*)(unsigned int)(uintptr_t)(lp), \
      0, 0)
#endif
#if defined(__HIP_DEVICE_COMPILE__) && __has_builtin(__builtin_amdgcn_tensor_load_to_lds) && __has_builtin(__builtin_amdgcn_s_wait_tensorcnt)
#define HAVE_TDM 1
#endif

// ---------------------------------------------------------------------------
// conv3x3 (SAME, NCHW) as implicit GEMM via v_wmma_f32_16x16x32_f16.
// Block = 128 thr (4 waves). Tile: 32 cout x 64 pixels. K = Cin*9, zero-padded.
// Wave w handles pixel sub-tile [p0+16w .. p0+16w+15], both cout sub-tiles.
// ---------------------------------------------------------------------------
__global__ void __launch_bounds__(128)
k_conv3x3_wmma(const float* __restrict__ x, const float* __restrict__ w,
               float* __restrict__ out, int Cin, int Cout, int H, int W) {
  const int HW  = H * W;
  const int K   = Cin * 9;
  const int p0  = blockIdx.x * 64;
  const int co0 = blockIdx.y * 32;
  const int n   = blockIdx.z;
  __shared__ __attribute__((aligned(16))) float As[32 * 32]; // weights [cout][k]
  __shared__ __attribute__((aligned(16))) float Bs[64 * 32]; // im2col  [pixel][k]
  const int tid  = threadIdx.x;
  const int wv   = tid >> 5;
  const int lane = tid & 31;
  const int half = lane >> 4;
  const int lm   = lane & 15;
  const bool kAligned = ((K & 31) == 0);
  v8f acc0 = {}, acc1 = {};
  for (int kc = 0; kc < K; kc += 32) {
    __builtin_prefetch(w + (size_t)(co0 + lane) * K + kc + 32, 0, 0);
    // ---- weights tile: async DMA to LDS when fully in-bounds ----
#if HAVE_ASYNC
    if (kAligned) {
      #pragma unroll
      for (int i = 0; i < 2; ++i) {                     // 2*128 lanes * 16B
        const int e = (i * 128 + tid) * 4;              // float idx in tile
        ASYNC_B128(w + (size_t)(co0 + (e >> 5)) * K + kc + (e & 31), &As[e]);
      }
    } else
#endif
    {
      for (int e = tid; e < 1024; e += 128) {
        const int k = kc + (e & 31);
        As[e] = (k < K) ? w[(size_t)(co0 + (e >> 5)) * K + k] : 0.f;
      }
    }
    // ---- im2col tile: gather with zero padding (sync) ----
    for (int e = tid; e < 2048; e += 128) {
      const int r = e >> 5, k = kc + (e & 31);
      float v = 0.f;
      if (k < K) {
        const int ci = k / 9, rr = k - ci * 9, kh = rr / 3, kw = rr - kh * 3;
        const int p  = p0 + r, oh = p / W, ow = p - oh * W;
        const int ih = oh + kh - 1, iw = ow + kw - 1;
        if (ih >= 0 && ih < H && iw >= 0 && iw < W)
          v = x[(((size_t)n * Cin + ci) * H + ih) * W + iw];
      }
      Bs[e] = v;
    }
#if HAVE_ASYNC
    if (kAligned) __builtin_amdgcn_s_wait_asynccnt(0);
#endif
    __syncthreads();
    // ---- fragments per ISA 16-bit layouts ----
    v16h a0, a1, b;
    #pragma unroll
    for (int j = 0; j < 16; ++j) {
      const int ka = ((j < 8) ? 0 : 16) + half * 8 + (j & 7);
      a0[j] = (_Float16)As[lm * 32 + ka];
      a1[j] = (_Float16)As[(16 + lm) * 32 + ka];
      b[j]  = (_Float16)Bs[(wv * 16 + lm) * 32 + ka];
    }
    acc0 = __builtin_amdgcn_wmma_f32_16x16x32_f16(false, a0, false, b,
                                                  (short)0, acc0, false, false);
    acc1 = __builtin_amdgcn_wmma_f32_16x16x32_f16(false, a1, false, b,
                                                  (short)0, acc1, false, false);
    __syncthreads();
  }
  // C/D layout: VGPR r -> M = half*8+r (cout), N = lane&15 (pixel)
  #pragma unroll
  for (int r = 0; r < 8; ++r) {
    const int m = half * 8 + r;
    const int p = p0 + wv * 16 + lm;
    out[((size_t)n * Cout + (co0 + m)) * HW + p]       = acc0[r];
    out[((size_t)n * Cout + (co0 + 16 + m)) * HW + p]  = acc1[r];
  }
}

// ---------------------------------------------------------------------------
// Generic GEMM: out[m][n] = sum_k A[m][k] * Wt[n][k] (+ bias[n]).
// Block = 128 thr (4 waves). Tile 64(M) x 64(N); wave w owns M rows 16w..16w+15
// and iterates 4 N sub-tiles (A-fragment reused). M % 64 == 0 for all calls.
// ---------------------------------------------------------------------------
__global__ void __launch_bounds__(128)
k_lin_wmma(const float* __restrict__ A, const float* __restrict__ Wt,
           const float* __restrict__ bias, float* __restrict__ out,
           int M, int K, int N) {
  const int m0 = blockIdx.x * 64;
  const int n0 = blockIdx.y * 64;
  __shared__ __attribute__((aligned(16))) float As[64 * 32];
  __shared__ __attribute__((aligned(16))) float Bs[64 * 32];
  const int tid  = threadIdx.x;
  const int wv   = tid >> 5;
  const int lane = tid & 31;
  const int half = lane >> 4;
  const int lm   = lane & 15;
  const bool kAligned = ((K & 31) == 0);
  const bool fullN    = kAligned && (n0 + 64 <= N);
  v8f acc[4] = {};
  for (int kc = 0; kc < K; kc += 32) {
#if HAVE_ASYNC
    if (kAligned) {
      #pragma unroll
      for (int i = 0; i < 4; ++i) {
        const int e = (i * 128 + tid) * 4;
        ASYNC_B128(A + (size_t)(m0 + (e >> 5)) * K + kc + (e & 31), &As[e]);
      }
    } else
#endif
    {
      for (int e = tid; e < 2048; e += 128) {
        const int k = kc + (e & 31);
        As[e] = (k < K) ? A[(size_t)(m0 + (e >> 5)) * K + k] : 0.f;
      }
    }
#if HAVE_ASYNC
    if (fullN) {
      #pragma unroll
      for (int i = 0; i < 4; ++i) {
        const int e = (i * 128 + tid) * 4;
        ASYNC_B128(Wt + (size_t)(n0 + (e >> 5)) * K + kc + (e & 31), &Bs[e]);
      }
    } else
#endif
    {
      for (int e = tid; e < 2048; e += 128) {
        const int r = e >> 5, k = kc + (e & 31);
        Bs[e] = (k < K && n0 + r < N) ? Wt[(size_t)(n0 + r) * K + k] : 0.f;
      }
    }
#if HAVE_ASYNC
    if (kAligned) __builtin_amdgcn_s_wait_asynccnt(0);
#endif
    __syncthreads();
    v16h a, b;
    #pragma unroll
    for (int j = 0; j < 16; ++j) {
      const int ka = ((j < 8) ? 0 : 16) + half * 8 + (j & 7);
      a[j] = (_Float16)As[(wv * 16 + lm) * 32 + ka];
    }
    #pragma unroll
    for (int nn = 0; nn < 4; ++nn) {
      if (n0 + nn * 16 < N) {                    // block-uniform guard
        #pragma unroll
        for (int j = 0; j < 16; ++j)
          b[j] = (_Float16)Bs[(nn * 16 + lm) * 32 + ((j < 8) ? 0 : 16) + half * 8 + (j & 7)];
        acc[nn] = __builtin_amdgcn_wmma_f32_16x16x32_f16(false, a, false, b,
                                                 (short)0, acc[nn], false, false);
      }
    }
    __syncthreads();
  }
  #pragma unroll
  for (int nn = 0; nn < 4; ++nn) {
    if (n0 + nn * 16 < N) {
      const float bv = bias ? bias[n0 + nn * 16 + lm] : 0.f;
      #pragma unroll
      for (int r = 0; r < 8; ++r) {
        const int m = wv * 16 + half * 8 + r;
        out[(size_t)(m0 + m) * N + (n0 + nn * 16 + lm)] = acc[nn][r] + bv;
      }
    }
  }
}

// ---------------------------------------------------------------------------
// Skip-connection copy into concat buffer via Tensor Data Mover.
// One block per (n,c) plane; TDM stages <=4096-float chunks into LDS, then
// the block fans out to the destination. Fallback: plain strided copy.
// ---------------------------------------------------------------------------
__global__ void __launch_bounds__(256)
k_copy_tdm(const float* __restrict__ src, float* __restrict__ dst,
           int Cs, int Cdst, int HW) {
  const int c = blockIdx.x % Cs;
  const int n = blockIdx.x / Cs;
  const float* sp = src + ((size_t)n * Cs + c) * HW;
  float*       dp = dst + ((size_t)n * Cdst + c) * HW;
#if HAVE_TDM
  __shared__ __attribute__((aligned(16))) float buf[4096];
  for (int off = 0; off < HW; off += 4096) {
    const int len = (HW - off < 4096) ? (HW - off) : 4096;
    if (threadIdx.x < 32) {   // one wave issues the DMA (TDM ignores EXEC)
      const unsigned long long ga = (unsigned long long)(uintptr_t)(sp + off);
      u32x4 g0 = {};
      g0[0] = 1u;                                           // count=1
      g0[1] = (unsigned int)(uintptr_t)buf;                 // lds_addr
      g0[2] = (unsigned int)(ga & 0xffffffffu);             // global_addr lo
      g0[3] = (unsigned int)((ga >> 32) & 0x01ffffffu) | (2u << 30); // +type=2
      i32x8 g1 = {};
      g1[0] = (int)(2u << 16);                              // data_size=4B
      g1[1] = (int)((unsigned)len << 16);                   // tensor_dim0 lo16
      g1[2] = (int)(((unsigned)len >> 16) & 0xffffu) | (1 << 16); // dim0hi|dim1lo
      g1[3] = (int)((unsigned)len << 16);                   // dim1 hi=0|tile_dim0
      g1[4] = 1;                                            // tile_dim1=1
      g1[5] = (int)(unsigned)len;                           // dim0_stride lo32
      i32x4 z4 = {};
#if __clang_major__ >= 23
      i32x8 z8 = {};
      __builtin_amdgcn_tensor_load_to_lds(g0, g1, z4, z4, z8, 0);
#else
      __builtin_amdgcn_tensor_load_to_lds(g0, g1, z4, z4, 0);
#endif
      __builtin_amdgcn_s_wait_tensorcnt(0);
    }
    __syncthreads();
    for (int i = threadIdx.x; i < len; i += 256) dp[off + i] = buf[i];
    __syncthreads();
  }
#else
  for (int i = threadIdx.x; i < HW; i += 256) dp[i] = sp[i];
#endif
}

// ---------------------------------------------------------------------------
// BatchNorm (training-mode) stats: one block per channel -> mean, inv_std
// ---------------------------------------------------------------------------
__global__ void k_bnstats(const float* __restrict__ x, float* __restrict__ stats,
                          int C, int HW) {
  const int c = blockIdx.x;
  __shared__ float s1[256], s2[256];
  float a = 0.f, b = 0.f;
  for (int idx = threadIdx.x; idx < 2 * HW; idx += 256) {
    const int n = idx / HW, p = idx - n * HW;
    const float v = x[(((size_t)n * C) + c) * HW + p];
    a += v; b += v * v;
  }
  s1[threadIdx.x] = a; s2[threadIdx.x] = b;
  __syncthreads();
  for (int o = 128; o > 0; o >>= 1) {
    if (threadIdx.x < o) { s1[threadIdx.x] += s1[threadIdx.x + o];
                           s2[threadIdx.x] += s2[threadIdx.x + o]; }
    __syncthreads();
  }
  if (threadIdx.x == 0) {
    const float cnt = (float)(2 * HW);
    const float mu  = s1[0] / cnt;
    const float var = s2[0] / cnt - mu * mu;
    stats[c]     = mu;
    stats[C + c] = rsqrtf(var + EPSV);
  }
}

__global__ void k_bn_relu(float* __restrict__ x, const float* __restrict__ stats,
                          const float* __restrict__ g, const float* __restrict__ b,
                          int C, int HW, int total) {
  const int idx = blockIdx.x * blockDim.x + threadIdx.x;
  if (idx >= total) return;
  const int c = (idx / HW) % C;
  const float v = (x[idx] - stats[c]) * stats[C + c] * g[c] + b[c];
  x[idx] = fmaxf(v, 0.f);
}

// 2x2 max pool, stride 2
__global__ void k_pool(const float* __restrict__ x, float* __restrict__ y,
                       int H, int W, int total) {
  const int idx = blockIdx.x * blockDim.x + threadIdx.x;
  if (idx >= total) return;
  const int Wo = W >> 1, Ho = H >> 1;
  const int wo = idx % Wo;
  const int ho = (idx / Wo) % Ho;
  const int nc = idx / (Wo * Ho);
  const float* xp = x + ((size_t)nc * H + 2 * ho) * W + 2 * wo;
  y[idx] = fmaxf(fmaxf(xp[0], xp[1]), fmaxf(xp[W], xp[W + 1]));
}

// LayerNorm over C, NCHW -> (B,L,C) row-major
__global__ void k_ln(const float* __restrict__ x, const float* __restrict__ g,
                     const float* __restrict__ b, float* __restrict__ xs,
                     int C, int L) {
  const int bl = blockIdx.x;
  const int n = bl / L, p = bl - n * L;
  __shared__ float red[256];
  const float* xp = x + (size_t)n * C * L + p;
  float s = 0.f;
  for (int c = threadIdx.x; c < C; c += 256) s += xp[(size_t)c * L];
  red[threadIdx.x] = s; __syncthreads();
  for (int o = 128; o > 0; o >>= 1) {
    if (threadIdx.x < o) red[threadIdx.x] += red[threadIdx.x + o];
    __syncthreads();
  }
  const float mu = red[0] / C;
  __syncthreads();
  float q = 0.f;
  for (int c = threadIdx.x; c < C; c += 256) {
    const float d = xp[(size_t)c * L] - mu; q += d * d;
  }
  red[threadIdx.x] = q; __syncthreads();
  for (int o = 128; o > 0; o >>= 1) {
    if (threadIdx.x < o) red[threadIdx.x] += red[threadIdx.x + o];
    __syncthreads();
  }
  const float istd = rsqrtf(red[0] / C + EPSV);
  for (int c = threadIdx.x; c < C; c += 256)
    xs[(size_t)bl * C + c] = (xp[(size_t)c * L] - mu) * istd * g[c] + b[c];
}

// Mamba recurrence: s = tanh(s @ A^T + xb_t), 16-wide state, sequential in L
__global__ void k_scan(const float* __restrict__ xB, const float* __restrict__ A,
                       float* __restrict__ st, int L) {
  const int b = blockIdx.x;
  const int i = threadIdx.x & 15;
  __shared__ float s[16];
  float Ar[16];
  #pragma unroll
  for (int j = 0; j < 16; ++j) Ar[j] = A[i * 16 + j];
  if (threadIdx.x < 16) s[i] = 0.f;
  __syncthreads();
  for (int l = 0; l < L; ++l) {
    float acc = xB[((size_t)b * L + l) * 16 + i];
    #pragma unroll
    for (int j = 0; j < 16; ++j) acc += Ar[j] * s[j];
    const float v = tanhf(acc);
    __syncthreads();
    if (threadIdx.x < 16) { s[i] = v; st[((size_t)b * L + l) * 16 + i] = v; }
    __syncthreads();
  }
}

__global__ void k_axpyD(float* __restrict__ y, const float* __restrict__ xin,
                        const float* __restrict__ D, int E, int total) {
  const int idx = blockIdx.x * blockDim.x + threadIdx.x;
  if (idx >= total) return;
  y[idx] += xin[idx] * D[idx % E];
}

// x (NCHW) += blc[(n*L+p)*C + c]   (residual add, layout swap)
__global__ void k_add_blc(const float* __restrict__ blc, float* __restrict__ x,
                          int C, int L, int total) {
  const int idx = blockIdx.x * blockDim.x + threadIdx.x;
  if (idx >= total) return;
  const int p = idx % L;
  const int c = (idx / L) % C;
  const int n = idx / (L * C);
  x[idx] += blc[((size_t)n * L + p) * C + c];
}

// ConvTranspose2d(k=2, s=2): writes Co channels at offset coff of concat dst
__global__ void k_upconv(const float* __restrict__ x, const float* __restrict__ uw,
                         const float* __restrict__ ub, float* __restrict__ dst,
                         int Ci, int Co, int H, int W, int Cdst, int coff, int total) {
  const int idx = blockIdx.x * blockDim.x + threadIdx.x;
  if (idx >= total) return;                 // total = 2*Co*H*W
  const int HW = H * W;
  const int p = idx % HW;
  const int o = (idx / HW) % Co;
  const int n = idx / (HW * Co);
  const int h = p / W, wq = p - h * W;
  float a00 = ub[o], a01 = ub[o], a10 = ub[o], a11 = ub[o];
  for (int i = 0; i < Ci; ++i) {
    const float xv = x[((size_t)n * Ci + i) * HW + p];
    const float* wp = uw + ((size_t)i * Co + o) * 4;   // (i,o,kh,kw)
    a00 += xv * wp[0]; a01 += xv * wp[1]; a10 += xv * wp[2]; a11 += xv * wp[3];
  }
  const size_t base = (((size_t)n * Cdst + coff + o) * (2 * H) + 2 * h) * (2 * W) + 2 * wq;
  dst[base] = a00; dst[base + 1] = a01;
  dst[base + 2 * W] = a10; dst[base + 2 * W + 1] = a11;
}

// final 1x1 conv: Cin=64 -> 1
__global__ void k_outc(const float* __restrict__ u, const float* __restrict__ w,
                       const float* __restrict__ b, float* __restrict__ out,
                       int C, int HW, int total) {
  const int idx = blockIdx.x * blockDim.x + threadIdx.x;
  if (idx >= total) return;                 // total = 2*HW
  const int p = idx % HW, n = idx / HW;
  float acc = b[0];
  for (int c = 0; c < C; ++c) acc += u[((size_t)n * C + c) * HW + p] * w[c];
  out[idx] = acc;
}

// ---------------------------------------------------------------------------
// Host orchestration
// ---------------------------------------------------------------------------
struct DCp { const float *w1,*g1,*b1,*w2,*g2,*b2; };
struct MBp { const float *A,*Bm,*Cm,*D,*ln_g,*ln_b,*in_w,*in_b,*out_w,*out_b; };

extern "C" void kernel_launch(void* const* d_in, const int* in_sizes, int n_in,
                              void* d_out, int out_size, void* d_ws, size_t ws_size,
                              hipStream_t stream) {
  (void)in_sizes; (void)n_in; (void)out_size; (void)ws_size;
  auto F = [&](int k) { return (const float*)d_in[k]; };
  auto getDC = [&](int i) { return DCp{F(i),F(i+1),F(i+2),F(i+3),F(i+4),F(i+5)}; };
  auto getMB = [&](int i) { return MBp{F(i),F(i+1),F(i+2),F(i+3),F(i+4),
                                       F(i+5),F(i+6),F(i+7),F(i+8),F(i+9)}; };
  const float* X = F(0);
  DCp inc = getDC(1);   MBp m1 = getMB(7);
  DCp dn1 = getDC(17);  MBp m2 = getMB(23);
  DCp dn2 = getDC(33);  MBp m3 = getMB(39);
  DCp dn3 = getDC(49);  MBp m4 = getMB(55);
  DCp dn4 = getDC(65);  MBp m5 = getMB(71);

  char* wp = (char*)d_ws;
  auto alloc = [&](size_t nfl) {
    float* p = (float*)wp;
    wp += ((nfl * sizeof(float) + 255) / 256) * 256;
    return p;
  };
  float* skip1 = alloc(2u * 64   * 128 * 128);
  float* skip2 = alloc(2u * 128  * 64  * 64);
  float* skip3 = alloc(2u * 256  * 32  * 32);
  float* skip4 = alloc(2u * 512  * 16  * 16);
  float* skip5 = alloc(2u * 1024 * 8   * 8);
  float* tmpA  = alloc(4194304);
  float* tmpB  = alloc(2097152);
  float* xs    = alloc(2097152);
  float* xin   = alloc(4194304);
  float* xB    = alloc(524288);
  float* st    = alloc(524288);
  float* yb    = alloc(4194304);
  float* decA  = alloc(2097152);
  float* decB  = alloc(2097152);
  float* stats = alloc(2048);

  auto conv3 = [&](const float* x, const float* w, float* y,
                   int Cin, int Cout, int H, int W) {
    dim3 g(H * W / 64, Cout / 32, 2);
    k_conv3x3_wmma<<<g, 128, 0, stream>>>(x, w, y, Cin, Cout, H, W);
  };
  auto bnrelu = [&](float* y, const float* g, const float* b, int C, int H, int W) {
    k_bnstats<<<C, 256, 0, stream>>>(y, stats, C, H * W);
    const int t = 2 * C * H * W;
    k_bn_relu<<<(t + 255) / 256, 256, 0, stream>>>(y, stats, g, b, C, H * W, t);
  };
  auto dconv = [&](const float* xi, const DCp& p, float* y,
                   int Cin, int Cout, int H, int W) {
    conv3(xi, p.w1, tmpB, Cin, Cout, H, W);
    bnrelu(tmpB, p.g1, p.b1, Cout, H, W);
    conv3(tmpB, p.w2, y, Cout, Cout, H, W);
    bnrelu(y, p.g2, p.b2, Cout, H, W);
  };
  auto lin = [&](const float* A, const float* Wt, const float* bias, float* o,
                 int M, int K, int N) {
    dim3 g(M / 64, (N + 63) / 64, 1);
    k_lin_wmma<<<g, 128, 0, stream>>>(A, Wt, bias, o, M, K, N);
  };
  auto mamba = [&](float* xio, const MBp& p, int C, int H, int W) {
    const int L = H * W, BL = 2 * L, E = 2 * C;
    k_ln<<<BL, 256, 0, stream>>>(xio, p.ln_g, p.ln_b, xs, C, L);
    lin(xs, p.in_w, p.in_b, xin, BL, C, E);
    lin(xin, p.Bm, nullptr, xB, BL, E, 16);
    k_scan<<<2, 32, 0, stream>>>(xB, p.A, st, L);
    lin(st, p.Cm, nullptr, yb, BL, 16, E);
    const int tE = BL * E;
    k_axpyD<<<(tE + 255) / 256, 256, 0, stream>>>(yb, xin, p.D, E, tE);
    lin(yb, p.out_w, p.out_b, xs, BL, E, C);
    const int tC = BL * C;
    k_add_blc<<<(tC + 255) / 256, 256, 0, stream>>>(xs, xio, C, L, tC);
  };
  auto pool = [&](const float* x, float* y, int C, int H, int W) {
    const int t = 2 * C * (H / 2) * (W / 2);
    k_pool<<<(t + 255) / 256, 256, 0, stream>>>(x, y, H, W, t);
  };
  auto up = [&](const float* xlow, const float* skip, int pb, float* y,
                int Ci, int Hs, int Ws) {
    const float* uw = F(pb); const float* ub = F(pb + 1);
    const DCp c = getDC(pb + 2);
    const int Co = Ci / 2, Cc = Ci, H2 = 2 * Hs, W2 = 2 * Ws, HW2 = H2 * W2;
    k_copy_tdm<<<2 * Co, 256, 0, stream>>>(skip, tmpA, Co, Cc, HW2);
    const int t2 = 2 * Co * Hs * Ws;
    k_upconv<<<(t2 + 255) / 256, 256, 0, stream>>>(xlow, uw, ub, tmpA,
                                                   Ci, Co, Hs, Ws, Cc, Co, t2);
    dconv(tmpA, c, y, Cc, Co, H2, W2);
  };

  // ---- encoder ----
  dconv(X, inc, skip1, 3, 64, 128, 128);          mamba(skip1, m1, 64, 128, 128);
  pool(skip1, tmpA, 64, 128, 128);
  dconv(tmpA, dn1, skip2, 64, 128, 64, 64);       mamba(skip2, m2, 128, 64, 64);
  pool(skip2, tmpA, 128, 64, 64);
  dconv(tmpA, dn2, skip3, 128, 256, 32, 32);      mamba(skip3, m3, 256, 32, 32);
  pool(skip3, tmpA, 256, 32, 32);
  dconv(tmpA, dn3, skip4, 256, 512, 16, 16);      mamba(skip4, m4, 512, 16, 16);
  pool(skip4, tmpA, 512, 16, 16);
  dconv(tmpA, dn4, skip5, 512, 1024, 8, 8);       mamba(skip5, m5, 1024, 8, 8);

  // ---- decoder ----
  up(skip5, skip4, 81,  decA, 1024, 8, 8);
  up(decA,  skip3, 89,  decB, 512, 16, 16);
  up(decB,  skip2, 97,  decA, 256, 32, 32);
  up(decA,  skip1, 105, decB, 128, 64, 64);

  const int to = 2 * 128 * 128;
  k_outc<<<(to + 255) / 256, 256, 0, stream>>>(decB, F(113), F(114),
                                               (float*)d_out, 64, 128 * 128, to);
}
